// TrajectoryEncoder_68049461838530
// MI455X (gfx1250) — compile-verified, matrix operand order checked by reading, not run on hardware
//
#include <hip/hip_runtime.h>
#include <hip/hip_bf16.h>

// ---------------------------------------------------------------------------
// TrajectoryEncoder for MI455X (gfx1250, wave32, WMMA).
// B=4096, T=2048, L=64, E=128, H=4, d=32, C=64, NL=2, FFN=256.
// Pass 0 converts all GEMM weights to bf16 in d_ws (run every call ->
// deterministic). All GEMMs use v_wmma_f32_16x16x32_bf16 with bf16 operands
// loaded as 16B vectors (ds_load_b128 / global_load_b128), f32 accumulate.
// d_out (B*L*E f32) doubles as the h buffer between kernels.
// ---------------------------------------------------------------------------

typedef __attribute__((ext_vector_type(16))) __bf16 v16bf;
typedef __attribute__((ext_vector_type(8)))  float  v8f;

__device__ __forceinline__ unsigned short f2bf(float x) {
    return __builtin_bit_cast(unsigned short, (__bf16)x);   // hw cvt, RNE
}
__device__ __forceinline__ __bf16 bfbits(unsigned short u) {
    return __builtin_bit_cast(__bf16, u);
}
__device__ __forceinline__ float geluf(float x) {
    return 0.5f * x * (1.0f + erff(x * 0.70710678118f));
}
__device__ __forceinline__ v8f vz8() { v8f z = {0,0,0,0,0,0,0,0}; return z; }

__device__ __forceinline__ v8f wmma_bf16(v16bf a, v16bf b, v8f c) {
    return __builtin_amdgcn_wmma_f32_16x16x32_bf16(false, a, false, b, (short)0, c, false, false);
}

// --- Fragment loaders -------------------------------------------------------
// 16-bit A/B per-lane k pattern (ISA 7.12.2): half = lane>>4, idx = lane&15;
// element e (=2v+s) -> k = (v>>2)*16 + half*8 + (v&3)*2 + s
// i.e. two contiguous runs of 8: [k0+half*8 .. +7] and [k0+16+half*8 .. +7].

// bf16 source (LDS or global), row-major: lane idx = A-row / B-out-col.
__device__ __forceinline__ v16bf frag_row_u16(const unsigned short* S, int ld, int row0, int k0) {
    int lane = threadIdx.x & 31, half = lane >> 4, m = lane & 15;
    const unsigned short* p = S + (size_t)(row0 + m) * ld + k0 + half * 8;
    v16bf f;
#pragma unroll
    for (int i = 0; i < 8; ++i) { f[i] = bfbits(p[i]); f[8 + i] = bfbits(p[16 + i]); }
    return f;
}
// A fragment from LDS f32 with on-the-fly bf16 conversion (hw cvt).
__device__ __forceinline__ v16bf frag_row_f32(const float* S, int ld, int row0, int k0) {
    int lane = threadIdx.x & 31, half = lane >> 4, m = lane & 15;
    const float* p = S + (row0 + m) * ld + k0 + half * 8;
    v16bf f;
#pragma unroll
    for (int i = 0; i < 8; ++i) { f[i] = (__bf16)p[i]; f[8 + i] = (__bf16)p[16 + i]; }
    return f;
}

// Row LayerNorm over 128 elements (eps 1e-5); optional bf16 mirror.
__device__ __forceinline__ void ln_row128(float* row, const float* g, const float* bb,
                                          unsigned short* bfrow) {
    float s = 0.f;
#pragma unroll 4
    for (int e = 0; e < 128; ++e) s += row[e];
    float mu = s * (1.0f / 128.0f);
    float v = 0.f;
#pragma unroll 4
    for (int e = 0; e < 128; ++e) { float d = row[e] - mu; v += d * d; }
    float inv = rsqrtf(v * (1.0f / 128.0f) + 1e-5f);
#pragma unroll 4
    for (int e = 0; e < 128; ++e) {
        float val = (row[e] - mu) * inv * g[e] + bb[e];
        row[e] = val;
        if (bfrow) bfrow[e] = f2bf(val);
    }
}

// ===========================================================================
// Kernel 0: f32 -> bf16 weight conversion into workspace.
// ===========================================================================
__global__ __launch_bounds__(256) void cvt_f32_bf16(const float* __restrict__ src,
                                                    unsigned short* __restrict__ dst, int n) {
    int i = blockIdx.x * 256 + threadIdx.x;
    if (i < n) dst[i] = f2bf(src[i]);
}

// ===========================================================================
// Kernel 1: segment pool -> conv(3x4) -> GroupNorm -> GELU -> proj (WMMA)
//           -> LayerNorm -> +PE   ==> d_out holds h0 (B,64,128) f32
// ===========================================================================
__global__ __launch_bounds__(256) void traj_stage1(
    const float* __restrict__ x, const unsigned char* __restrict__ mask,
    const float* __restrict__ cw, const float* __restrict__ cb,
    const float* __restrict__ gng, const float* __restrict__ gnb,
    const unsigned short* __restrict__ pwbf, const float* __restrict__ pb,
    const float* __restrict__ lng, const float* __restrict__ lnb,
    float* __restrict__ out)
{
    __shared__ float s_sum[64][4];            // xyz sums + count per bin
    __shared__ float s_pool[64][4];           // filled bin means + ones channel
    __shared__ float s_cv[64][64];            // conv out, [l][c]
    __shared__ unsigned short s_act[64][64];  // bf16 activations (A operand)
    __shared__ float s_o[64][128];            // projected output
    __shared__ float s_red[256];
    __shared__ float s_red2[256];
    __shared__ int   s_Ti;

    const int b = blockIdx.x;
    const int tid = threadIdx.x;

    // ---- valid-token count (prefix mask => rank(t)==t) ----
    int cnt = 0;
    for (int t = tid; t < 2048; t += 256) cnt += mask[(size_t)b * 2048 + t] ? 1 : 0;
    s_red[tid] = (float)cnt;
    __syncthreads();
    for (int off = 128; off; off >>= 1) {
        if (tid < off) s_red[tid] += s_red[tid + off];
        __syncthreads();
    }
    if (tid == 0) s_Ti = (int)s_red[0];
    if (tid < 64) { s_sum[tid][0] = 0.f; s_sum[tid][1] = 0.f; s_sum[tid][2] = 0.f; s_sum[tid][3] = 0.f; }
    __syncthreads();
    const int Ti = s_Ti;

    // ---- bin accumulation (LDS float atomics) ----
    for (int t = tid; t < Ti; t += 256) {
        int bin = (t * 64) / Ti;
        const float* xp = x + ((size_t)b * 2048 + t) * 4;
        atomicAdd(&s_sum[bin][0], xp[0]);
        atomicAdd(&s_sum[bin][1], xp[1]);
        atomicAdd(&s_sum[bin][2], xp[2]);
        atomicAdd(&s_sum[bin][3], 1.0f);
    }
    __syncthreads();

    // ---- means + forward/backward fill (serial, 64 bins) ----
    if (tid == 0) {
        int first = 0;
        for (int k = 0; k < 64; ++k) if (s_sum[k][3] > 0.f) { first = k; break; }
        int run = -1;
        for (int k = 0; k < 64; ++k) {
            if (s_sum[k][3] > 0.f) run = k;
            int src = (run >= 0) ? run : first;
            if (Ti > 0) {
                float c = s_sum[src][3];
                s_pool[k][0] = s_sum[src][0] / c;
                s_pool[k][1] = s_sum[src][1] / c;
                s_pool[k][2] = s_sum[src][2] / c;
            } else {
                const float* xp = x + (size_t)b * 2048 * 4;
                s_pool[k][0] = xp[0]; s_pool[k][1] = xp[1]; s_pool[k][2] = xp[2];
            }
            s_pool[k][3] = 1.0f;
        }
    }
    __syncthreads();

    // ---- conv: out[c][l] stored as s_cv[l][c] ----
    for (int i = tid; i < 4096; i += 256) {
        int c = i >> 6, l = i & 63;
        float acc = cb[c];
#pragma unroll
        for (int kh = 0; kh < 3; ++kh) {
            int ll = l + kh - 1;
            if (ll < 0 || ll > 63) continue;
            const float* wr = cw + c * 12 + kh * 4;
            acc += s_pool[ll][0] * wr[0] + s_pool[ll][1] * wr[1]
                 + s_pool[ll][2] * wr[2] + s_pool[ll][3] * wr[3];
        }
        s_cv[l][c] = acc;
    }
    __syncthreads();

    // ---- GroupNorm(1 group over C*L) + GELU -> bf16 ----
    float ls = 0.f, lq = 0.f;
    for (int i = tid; i < 4096; i += 256) {
        float v = ((const float*)s_cv)[i];
        ls += v; lq += v * v;
    }
    s_red[tid] = ls; s_red2[tid] = lq;
    __syncthreads();
    for (int off = 128; off; off >>= 1) {
        if (tid < off) { s_red[tid] += s_red[tid + off]; s_red2[tid] += s_red2[tid + off]; }
        __syncthreads();
    }
    float mu = s_red[0] * (1.0f / 4096.0f);
    float var = s_red2[0] * (1.0f / 4096.0f) - mu * mu;
    float inv = rsqrtf(var + 1e-5f);
    for (int i = tid; i < 4096; i += 256) {
        int l = i >> 6, c = i & 63;
        float v = (s_cv[l][c] - mu) * inv * gng[c] + gnb[c];
        s_act[l][c] = f2bf(geluf(v));
    }
    __syncthreads();

    // ---- proj: (64x64) @ (64x128) via WMMA; proj_w bf16 is (E=128, C=64) ----
    const int w = tid >> 5, lane = tid & 31, half = lane >> 4, n = lane & 15;
#pragma unroll
    for (int i = 0; i < 4; ++i) {
        int tt = w * 4 + i;             // 0..31
        int mi = tt & 3, ni = tt >> 2;  // M tiles 4, N tiles 8
        v8f acc = vz8();
#pragma unroll
        for (int k0 = 0; k0 < 64; k0 += 32) {
            v16bf a = frag_row_u16(&s_act[0][0], 64, mi * 16, k0);
            v16bf bb = frag_row_u16(pwbf, 64, ni * 16, k0);
            acc = wmma_bf16(a, bb, acc);
        }
#pragma unroll
        for (int r = 0; r < 8; ++r) {
            int row = mi * 16 + r + half * 8, col = ni * 16 + n;
            s_o[row][col] = acc[r] + pb[col];
        }
    }
    __syncthreads();

    // ---- LayerNorm then + sinusoidal PE, write to d_out ----
    if (tid < 64) ln_row128(&s_o[tid][0], lng, lnb, nullptr);
    __syncthreads();
    for (int i = tid; i < 8192; i += 256) {
        int l = i >> 7, e = i & 127;
        int ii = e >> 1;
        float freq = __expf(-(float)(2 * ii) * (9.210340372f / 128.0f));
        float ang = (float)l * freq;
        float pe = (e & 1) ? __cosf(ang) : __sinf(ang);
        out[((size_t)b * 64 + l) * 128 + e] = s_o[l][e] + pe;
    }
}

// ===========================================================================
// Kernel 2: NL=2 post-norm transformer layers, one workgroup per batch elem.
// Weights come in as bf16 (converted once by kernel 0).
// ===========================================================================
__global__ __launch_bounds__(256) void traj_stage2(
    float* __restrict__ hio,
    const unsigned short* __restrict__ qkvW0, const float* __restrict__ qkvB0,
    const unsigned short* __restrict__ outW0, const float* __restrict__ outB0,
    const float* __restrict__ ln1g0, const float* __restrict__ ln1b0,
    const unsigned short* __restrict__ w10,   const float* __restrict__ b10,
    const unsigned short* __restrict__ w20,   const float* __restrict__ b20,
    const float* __restrict__ ln2g0, const float* __restrict__ ln2b0)
{
    __shared__ float          s_h[64][128];   // f32 hidden (residual stream)
    __shared__ unsigned short s_bf[64][128];  // bf16: attn output o / LN1(h)
    __shared__ unsigned short s_qk[64][64];   // per-head q|k -> attn; FFN f-chunk
    __shared__ unsigned short s_vT[32][64];   // per-head v, TRANSPOSED [dim][key]
    __shared__ float          s_rmax[64][2];
    __shared__ float          s_rsum[64][2];

    const int b = blockIdx.x, tid = threadIdx.x;
    const int w = tid >> 5, lane = tid & 31, half = lane >> 4, n = lane & 15;

    float* hptr = hio + (size_t)b * 64 * 128;
    for (int i = tid; i < 8192; i += 256) ((float*)s_h)[i] = hptr[i];
    __syncthreads();

    for (int l = 0; l < 2; ++l) {
        const unsigned short* qW = qkvW0 + (size_t)l * 384 * 128;
        const float* qB = qkvB0 + (size_t)l * 384;
        const unsigned short* oW = outW0 + (size_t)l * 128 * 128;
        const float* oB = outB0 + (size_t)l * 128;
        const float* g1 = ln1g0 + (size_t)l * 128;
        const float* h1 = ln1b0 + (size_t)l * 128;
        const unsigned short* W1 = w10 + (size_t)l * 256 * 128;
        const float* B1 = b10 + (size_t)l * 256;
        const unsigned short* W2 = w20 + (size_t)l * 128 * 256;
        const float* B2 = b20 + (size_t)l * 128;
        const float* g2 = ln2g0 + (size_t)l * 128;
        const float* h2 = ln2b0 + (size_t)l * 128;
        __builtin_prefetch(qW, 0, 1);
        __builtin_prefetch(W1, 0, 1);
        __builtin_prefetch(W2, 0, 1);

        // ================= attention (per head) =================
        for (int hd = 0; hd < 4; ++hd) {
            // ---- q,k,v for this head: 3 mats x (M4 x N2) = 24 tiles / 8 waves ----
#pragma unroll
            for (int i = 0; i < 3; ++i) {
                int tt = w * 3 + i;            // 0..23
                int mat = tt >> 3;             // 0=q 1=k 2=v
                int sub = tt & 7;
                int mi = sub & 3, ni = sub >> 2;       // ni 0..1
                int colbase = mat * 128 + hd * 32 + ni * 16;   // row of qkv_w
                v8f acc = vz8();
#pragma unroll
                for (int k0 = 0; k0 < 128; k0 += 32) {
                    v16bf a = frag_row_f32(&s_h[0][0], 128, mi * 16, k0);
                    v16bf bb = frag_row_u16(qW, 128, colbase, k0);
                    acc = wmma_bf16(a, bb, acc);
                }
                float sc = (mat == 0) ? 0.176776695f : 1.0f;   // q * 1/sqrt(32)
#pragma unroll
                for (int r = 0; r < 8; ++r) {
                    int row = mi * 16 + r + half * 8;
                    int c = ni * 16 + n;
                    unsigned short bv = f2bf((acc[r] + qB[colbase + n]) * sc);
                    if (mat == 0)      s_qk[row][c] = bv;       // q cols 0..31
                    else if (mat == 1) s_qk[row][32 + c] = bv;  // k cols 32..63
                    else               s_vT[c][row] = bv;       // v transposed
                }
            }
            __syncthreads();

            // ---- scores: q(64x32) @ k^T(32x64). wave: mi=w&3, col-half nh=w>>2 ----
            int mi = w & 3, nh = w >> 2;
            v8f sc0, sc1;
            {
                v16bf a = frag_row_u16(&s_qk[0][0], 64, mi * 16, 0);     // q cols 0..31
                v16bf bA = frag_row_u16(&s_qk[0][0], 64, nh * 32, 32);   // k rows (keys)
                v16bf bB = frag_row_u16(&s_qk[0][0], 64, nh * 32 + 16, 32);
                sc0 = wmma_bf16(a, bA, vz8());
                sc1 = wmma_bf16(a, bB, vz8());
            }
            // ---- softmax over 64 keys, split across 2 waves per row-group ----
#pragma unroll
            for (int r = 0; r < 8; ++r) {
                float m = fmaxf(sc0[r], sc1[r]);
                m = fmaxf(m, __shfl_xor(m, 1));
                m = fmaxf(m, __shfl_xor(m, 2));
                m = fmaxf(m, __shfl_xor(m, 4));
                m = fmaxf(m, __shfl_xor(m, 8));
                if (n == 0) s_rmax[mi * 16 + r + half * 8][nh] = m;
            }
            __syncthreads();
#pragma unroll
            for (int r = 0; r < 8; ++r) {
                int row = mi * 16 + r + half * 8;
                float M = fmaxf(s_rmax[row][0], s_rmax[row][1]);
                sc0[r] = __expf(sc0[r] - M);
                sc1[r] = __expf(sc1[r] - M);
                float s = sc0[r] + sc1[r];
                s += __shfl_xor(s, 1);
                s += __shfl_xor(s, 2);
                s += __shfl_xor(s, 4);
                s += __shfl_xor(s, 8);
                if (n == 0) s_rsum[row][nh] = s;
            }
            __syncthreads();
#pragma unroll
            for (int r = 0; r < 8; ++r) {
                int row = mi * 16 + r + half * 8;
                float is = 1.0f / (s_rsum[row][0] + s_rsum[row][1]);
                s_qk[row][nh * 32 + n]      = f2bf(sc0[r] * is);   // overlay q|k
                s_qk[row][nh * 32 + 16 + n] = f2bf(sc1[r] * is);
            }
            __syncthreads();

            // ---- o_hd = attn(64x64) @ v(64x32): 8 tiles / 8 waves ----
            {
                int mi2 = w & 3, ni2 = w >> 2;    // ni2 0..1
                v8f oacc = vz8();
#pragma unroll
                for (int k0 = 0; k0 < 64; k0 += 32) {
                    v16bf a = frag_row_u16(&s_qk[0][0], 64, mi2 * 16, k0);
                    // B[k=key][n=dim] = vT[dim][key]: row pattern on s_vT (ld 64)
                    v16bf bb = frag_row_u16(&s_vT[0][0], 64, ni2 * 16, k0);
                    oacc = wmma_bf16(a, bb, oacc);
                }
#pragma unroll
                for (int r = 0; r < 8; ++r) {
                    int row = mi2 * 16 + r + half * 8;
                    s_bf[row][hd * 32 + ni2 * 16 + n] = f2bf(oacc[r]);
                }
            }
            __syncthreads();
        } // heads

        // ---- out-proj (64x128)@(128x128) + residual ----
#pragma unroll
        for (int i = 0; i < 4; ++i) {
            int tt = w * 4 + i;
            int mi = tt & 3, ni = tt >> 2;   // ni 0..7
            v8f acc = vz8();
#pragma unroll
            for (int k0 = 0; k0 < 128; k0 += 32) {
                v16bf a = frag_row_u16(&s_bf[0][0], 128, mi * 16, k0);
                v16bf bb = frag_row_u16(oW, 128, ni * 16, k0);
                acc = wmma_bf16(a, bb, acc);
            }
#pragma unroll
            for (int r = 0; r < 8; ++r) {
                int row = mi * 16 + r + half * 8, col = ni * 16 + n;
                s_h[row][col] += acc[r] + oB[col];
            }
        }
        __syncthreads();
        if (tid < 64) ln_row128(&s_h[tid][0], g1, h1, &s_bf[tid][0]);  // LN1 + bf16 mirror
        __syncthreads();

        // ---- FFN: out = gelu(h@W1^T+b1)@W2^T + b2, chunked over 4x64 cols ----
        v8f facc[4];
#pragma unroll
        for (int i = 0; i < 4; ++i) facc[i] = vz8();
        for (int cc = 0; cc < 4; ++cc) {
            __syncthreads();  // prior reads of s_qk complete
            // f chunk: (64x128)@(128x64) -> gelu -> bf16 in s_qk
#pragma unroll
            for (int i = 0; i < 2; ++i) {
                int tt = w * 2 + i;
                int mi = tt & 3, ni = tt >> 2;   // ni 0..3
                v8f acc = vz8();
#pragma unroll
                for (int k0 = 0; k0 < 128; k0 += 32) {
                    v16bf a = frag_row_u16(&s_bf[0][0], 128, mi * 16, k0);
                    v16bf bb = frag_row_u16(W1, 128, cc * 64 + ni * 16, k0);
                    acc = wmma_bf16(a, bb, acc);
                }
#pragma unroll
                for (int r = 0; r < 8; ++r) {
                    int row = mi * 16 + r + half * 8;
                    int jc = cc * 64 + ni * 16 + n;
                    s_qk[row][ni * 16 + n] = f2bf(geluf(acc[r] + B1[jc]));
                }
            }
            __syncthreads();
            // accumulate out += f_cc @ W2^T[:, cc*64 : cc*64+64]
#pragma unroll
            for (int i = 0; i < 4; ++i) {
                int tt = w * 4 + i;
                int mi = tt & 3, ni = tt >> 2;   // ni 0..7
#pragma unroll
                for (int k0 = 0; k0 < 64; k0 += 32) {
                    v16bf a = frag_row_u16(&s_qk[0][0], 64, mi * 16, k0);
                    v16bf bb = frag_row_u16(W2, 256, ni * 16, cc * 64 + k0);
                    facc[i] = wmma_bf16(a, bb, facc[i]);
                }
            }
        }
        __syncthreads();
#pragma unroll
        for (int i = 0; i < 4; ++i) {
            int tt = w * 4 + i;
            int mi = tt & 3, ni = tt >> 2;
#pragma unroll
            for (int r = 0; r < 8; ++r) {
                int row = mi * 16 + r + half * 8, col = ni * 16 + n;
                s_h[row][col] += facc[i][r] + B2[col];
            }
        }
        __syncthreads();
        if (tid < 64) ln_row128(&s_h[tid][0], g2, h2, nullptr);        // LN2
        __syncthreads();
    } // layers

    for (int i = tid; i < 8192; i += 256) hptr[i] = ((const float*)s_h)[i];
}

// ===========================================================================
extern "C" void kernel_launch(void* const* d_in, const int* in_sizes, int n_in,
                              void* d_out, int out_size, void* d_ws, size_t ws_size,
                              hipStream_t stream) {
    const float* x       = (const float*)d_in[0];
    const float* conv_w  = (const float*)d_in[1];
    const float* conv_b  = (const float*)d_in[2];
    const float* gn_g    = (const float*)d_in[3];
    const float* gn_b    = (const float*)d_in[4];
    const float* proj_w  = (const float*)d_in[5];
    const float* proj_b  = (const float*)d_in[6];
    const float* ln_g    = (const float*)d_in[7];
    const float* ln_b    = (const float*)d_in[8];
    const float* qkv_w   = (const float*)d_in[9];
    const float* qkv_b   = (const float*)d_in[10];
    const float* out_w   = (const float*)d_in[11];
    const float* out_b   = (const float*)d_in[12];
    const float* ln1_g   = (const float*)d_in[13];
    const float* ln1_b   = (const float*)d_in[14];
    const float* w1      = (const float*)d_in[15];
    const float* b1      = (const float*)d_in[16];
    const float* w2      = (const float*)d_in[17];
    const float* b2      = (const float*)d_in[18];
    const float* ln2_g   = (const float*)d_in[19];
    const float* ln2_b   = (const float*)d_in[20];
    const unsigned char* pad_mask = (const unsigned char*)d_in[21];

    float* out = (float*)d_out;
    (void)in_sizes; (void)n_in; (void)out_size; (void)ws_size;

    // bf16 weight scratch layout (elements):
    //   proj 8192 | qkv 98304 | out 32768 | w1 65536 | w2 65536   (= 540,672 B)
    unsigned short* wb   = (unsigned short*)d_ws;
    unsigned short* projB = wb;
    unsigned short* qkvB16 = wb + 8192;
    unsigned short* outB16 = wb + 8192 + 98304;
    unsigned short* w1B16  = wb + 8192 + 98304 + 32768;
    unsigned short* w2B16  = wb + 8192 + 98304 + 32768 + 65536;

    cvt_f32_bf16<<<dim3((8192  + 255) / 256), dim3(256), 0, stream>>>(proj_w, projB, 8192);
    cvt_f32_bf16<<<dim3((98304 + 255) / 256), dim3(256), 0, stream>>>(qkv_w, qkvB16, 98304);
    cvt_f32_bf16<<<dim3((32768 + 255) / 256), dim3(256), 0, stream>>>(out_w, outB16, 32768);
    cvt_f32_bf16<<<dim3((65536 + 255) / 256), dim3(256), 0, stream>>>(w1, w1B16, 65536);
    cvt_f32_bf16<<<dim3((65536 + 255) / 256), dim3(256), 0, stream>>>(w2, w2B16, 65536);

    traj_stage1<<<dim3(4096), dim3(256), 0, stream>>>(
        x, pad_mask, conv_w, conv_b, gn_g, gn_b, projB, proj_b, ln_g, ln_b, out);

    traj_stage2<<<dim3(4096), dim3(256), 0, stream>>>(
        out, qkvB16, qkv_b, outB16, out_b, ln1_g, ln1_b, w1B16, b1, w2B16, b2, ln2_g, ln2_b);
}
